// SCnet_60069412602441
// MI455X (gfx1250) — matrix operand order repaired and checked by reading, not compiled, not run on hardware
//
#include <hip/hip_runtime.h>
#include <hip/hip_bf16.h>
#include <math.h>

// ---------------------------------------------------------------------------
// Sizes (compile-time constants from the reference)
// ---------------------------------------------------------------------------
#define NLAYERS 5
#define EMB     300
#define KPAD    320          // EMB padded to multiple of 32 (WMMA K-chunk)
#define NP      304          // EMB padded to multiple of 16 (WMMA N-tile)
#define NNODES  50000
#define NEDGES  200000
#define NMSG    (NEDGES + NNODES)   // 250000
#define ROWS    80                  // messages per block: 250000/80 = 3125 exact
#define RT      (ROWS / 16)         // 5 row tiles per block
#define NB      1000
#define NPER    50
#define DK      256

typedef __attribute__((ext_vector_type(16))) __bf16 v16bf;
typedef __attribute__((ext_vector_type(8)))  __bf16 v8bf;
typedef __attribute__((ext_vector_type(8)))  float  v8f;

// ---------------------------------------------------------------------------
// Convert mlp_w[l] (fp32, [300][300]) -> bf16 padded [304][320] per layer
// ---------------------------------------------------------------------------
__global__ void cvt_w_kernel(const float* __restrict__ mlp_w,
                             unsigned short* __restrict__ wbf_raw) {
  int idx = blockIdx.x * blockDim.x + threadIdx.x;
  if (idx >= NLAYERS * NP * KPAD) return;
  int l = idx / (NP * KPAD);
  int r = idx % (NP * KPAD);
  int n = r / KPAD, k = r % KPAD;
  float v = (n < EMB && k < EMB) ? mlp_w[(size_t)l * EMB * EMB + n * EMB + k] : 0.f;
  ((__bf16*)wbf_raw)[idx] = (__bf16)v;
}

// ---------------------------------------------------------------------------
// h = x_emb[x_type] + x_feat @ x_weight        [50000, 300]
// ---------------------------------------------------------------------------
__global__ void init_h_kernel(const int* __restrict__ x_type,
                              const float* __restrict__ x_feat,
                              const float* __restrict__ x_emb,
                              const float* __restrict__ x_weight,
                              float* __restrict__ h) {
  int idx = blockIdx.x * blockDim.x + threadIdx.x;
  if (idx >= NNODES * EMB) return;
  int n = idx / EMB, c = idx % EMB;
  float s = x_emb[x_type[n] * EMB + c];
#pragma unroll
  for (int p = 0; p < 16; ++p) s += x_feat[n * 16 + p] * x_weight[p * EMB + c];
  h[idx] = s;
}

// ---------------------------------------------------------------------------
// Message GEMM + scatter.  One block = 80 messages x 304 out channels.
// A (80x320 bf16) staged in LDS; each wave loads one B fragment per K-chunk
// and reuses it across 5 row-tiles (5 WMMAs per B load -> 5x less L2 weight
// traffic).  Accumulate f32 via v_wmma_f32_16x16x32_bf16, fuse bias+relu,
// scatter with global f32 atomics.
// ---------------------------------------------------------------------------
__global__ __launch_bounds__(128)
void msg_gemm_scatter_kernel(const float* __restrict__ h,
                             float* __restrict__ hnew,
                             const unsigned short* __restrict__ wbf_raw, // [304][320] bf16
                             const float* __restrict__ bias,             // [300]
                             const int* __restrict__ edge_index,         // [2*E]
                             const float* __restrict__ edge_attr,        // [E*2]
                             const float* __restrict__ ew0,              // [300]
                             const float* __restrict__ ew1v) {           // [300]
  __shared__ __bf16 At[ROWS][KPAD];    // 50 KB
  __shared__ int dst[ROWS];
  const __bf16* __restrict__ wb = (const __bf16*)wbf_raw;

  const int tid = threadIdx.x;
  const int m0  = blockIdx.x * ROWS;

  // ---- stage A tile: gather h[row] + rank-2 edge feature, cvt to bf16 ----
  // 16 threads per row, 8 rows in flight, 10 row-iterations.
  for (int r = tid >> 4; r < ROWS; r += 8) {
    int mid = m0 + r;
    int src, d; float ea0, ea1;
    if (mid < NEDGES) {
      src = edge_index[mid]; d = edge_index[NEDGES + mid];
      ea0 = edge_attr[2 * mid]; ea1 = edge_attr[2 * mid + 1];
    } else {                     // self loop: attr = (0, 4)
      int node = mid - NEDGES; src = node; d = node; ea0 = 0.f; ea1 = 4.f;
    }
    if ((tid & 15) == 0) dst[r] = d;
    const float* hs = h + (size_t)src * EMB;
    for (int c = tid & 15; c < KPAD; c += 16) {
      float v = 0.f;
      if (c < EMB) v = hs[c] + ea0 * ew0[c] + ea1 * ew1v[c];
      At[r][c] = (__bf16)v;
    }
  }
  __syncthreads();

  const int wave    = tid >> 5;       // 0..3
  const int lane    = tid & 31;
  const int lo      = lane & 15;
  const bool hiHalf = lane >= 16;

  for (int nt = wave; nt < NP / 16; nt += 4) {   // 19 column tiles
    const int n = nt * 16 + lo;                  // this lane's output column
    v8f acc[RT];
#pragma unroll
    for (int rt = 0; rt < RT; ++rt) acc[rt] = (v8f){};

    for (int kc = 0; kc < KPAD; kc += 32) {      // 10 K-chunks
      // B fragment (32x16 bf16): 16 consecutive K values of column n,
      // loaded once and reused for all 5 row tiles.
      int   kb = kc + (hiHalf ? 16 : 0);
      v16bf b  = *(const v16bf*)&wb[(size_t)n * KPAD + kb];
      int   ka = kc + (hiHalf ? 16 : 0);
#pragma unroll
      for (int rt = 0; rt < RT; ++rt) {
        // A fragment (16x32 bf16): low lanes K{kc..+15},
        //                          high lanes K{kc+16..+31}
        v8bf alo = *(const v8bf*)&At[rt * 16 + lo][ka];
        v8bf ahi = *(const v8bf*)&At[rt * 16 + lo][ka + 8];
        v16bf a;
#pragma unroll
        for (int i = 0; i < 8; ++i) { a[i] = alo[i]; a[i + 8] = ahi[i]; }
        acc[rt] = __builtin_amdgcn_wmma_f32_16x16x32_bf16(
                      false, a, false, b, (short)0, acc[rt], false, false);
      }
    }
    // ---- fuse bias + relu, scatter-add into hnew[col] ----
    if (n < EMB) {
      float bn = bias[n];
#pragma unroll
      for (int rt = 0; rt < RT; ++rt) {
#pragma unroll
        for (int rreg = 0; rreg < 8; ++rreg) {
          int   mrow = rt * 16 + (hiHalf ? (rreg + 8) : rreg);
          float v    = acc[rt][rreg] + bn;
          v = v > 0.f ? v : 0.f;
          atomicAdd(&hnew[(size_t)dst[mrow] * EMB + n], v);
        }
      }
    }
  }
}

// ---------------------------------------------------------------------------
// BatchNorm (eval) + optional ReLU:  h = (hnew-rm)*rsqrt(rv+eps)*g + b
// ---------------------------------------------------------------------------
__global__ void bn_kernel(const float* __restrict__ hin,
                          float* __restrict__ hout,
                          const float* __restrict__ g,  const float* __restrict__ bb,
                          const float* __restrict__ rm, const float* __restrict__ rv,
                          int do_relu) {
  int idx = blockIdx.x * blockDim.x + threadIdx.x;
  if (idx >= NNODES * EMB) return;
  int c = idx % EMB;
  float v = (hin[idx] - rm[c]) * rsqrtf(rv[c] + 1e-5f) * g[c] + bb[c];
  if (do_relu) v = fmaxf(v, 0.f);
  hout[idx] = v;
}

// ---------------------------------------------------------------------------
// Fused head: one block per graph. Uses the algebraic collapse
//   scores_i = (Q@Wk) . h_i ,  softmax@V = (sum a_i h_i) @ Wv^T
// so all GEMVs are over 300/256 vectors — pure VALU, trivially cheap.
// ---------------------------------------------------------------------------
__global__ __launch_bounds__(256)
void head_kernel(const float* __restrict__ h,
                 const int*   __restrict__ metal_type,
                 const float* __restrict__ metal_feat,
                 const float* __restrict__ me1_w, const float* __restrict__ me1_b,
                 const float* __restrict__ me2,
                 const float* __restrict__ wq, const float* __restrict__ wk,
                 const float* __restrict__ wv,
                 const float* __restrict__ ml_w, const float* __restrict__ ml_b,
                 const float* __restrict__ lg_w, const float* __restrict__ lg_b,
                 const float* __restrict__ ph1_w, const float* __restrict__ ph1_b,
                 const float* __restrict__ ph2_w, const float* __restrict__ ph2_b,
                 float* __restrict__ out) {
  __shared__ float mf_s[EMB], q_s[DK], qk_s[EMB], gm_s[EMB], hbar[EMB];
  __shared__ float sc[64], feat[DK], red[256];
  __shared__ float ssum;
  const int b = blockIdx.x;
  const int t = threadIdx.x;
  const float* hb = h + (size_t)b * NPER * EMB;

  // metal feature: Linear(17->300) + Embedding
  for (int c = t; c < EMB; c += 256) {
    float s = me1_b[c] + me2[metal_type[b] * EMB + c];
#pragma unroll
    for (int j = 0; j < 17; ++j) s += metal_feat[b * 17 + j] * me1_w[c * 17 + j];
    mf_s[c] = s;
  }
  // graph mean pool
  for (int c = t; c < EMB; c += 256) {
    float s = 0.f;
    for (int i = 0; i < NPER; ++i) s += hb[i * EMB + c];
    gm_s[c] = s * (1.f / NPER);
  }
  __syncthreads();

  // Q = mf@wq^T ; metal2 = relu(mf@ml_w^T + ml_b)   (t indexes d < 256)
  float m2;
  {
    float q = 0.f, m = ml_b[t];
    for (int c = 0; c < EMB; ++c) {
      float x = mf_s[c];
      q += x * wq[t * EMB + c];
      m += x * ml_w[t * EMB + c];
    }
    q_s[t] = q;
    m2 = m > 0.f ? m : 0.f;
  }
  __syncthreads();

  // qk = Q @ wk   (300-vector)
  for (int c = t; c < EMB; c += 256) {
    float s = 0.f;
    for (int d = 0; d < DK; ++d) s += q_s[d] * wk[d * EMB + c];
    qk_s[c] = s;
  }
  __syncthreads();

  // scores + softmax over the 50 nodes
  if (t < NPER) {
    float s = 0.f;
    for (int c = 0; c < EMB; ++c) s += qk_s[c] * hb[t * EMB + c];
    sc[t] = s * (1.f / 16.f);            // /sqrt(256)
  }
  __syncthreads();
  if (t == 0) {
    float mx = sc[0];
    for (int i = 1; i < NPER; ++i) mx = fmaxf(mx, sc[i]);
    float su = 0.f;
    for (int i = 0; i < NPER; ++i) { sc[i] = __expf(sc[i] - mx); su += sc[i]; }
    ssum = su;
  }
  __syncthreads();

  // hbar = softmax-weighted node mean
  for (int c = t; c < EMB; c += 256) {
    float s = 0.f;
    for (int i = 0; i < NPER; ++i) s += sc[i] * hb[i * EMB + c];
    hbar[c] = s / ssum;
  }
  __syncthreads();

  // feature = relu(hbar@wv^T) + metal2 + relu(gm@lg_w^T + lg_b)
  {
    float av = 0.f, lg = lg_b[t];
    for (int c = 0; c < EMB; ++c) {
      av += hbar[c] * wv[t * EMB + c];
      lg += gm_s[c] * lg_w[t * EMB + c];
    }
    av = av > 0.f ? av : 0.f;
    lg = lg > 0.f ? lg : 0.f;
    feat[t] = av + m2 + lg;
  }
  __syncthreads();

  // out = softplus(feature@ph1^T + b1) @ ph2^T + b2
  float p = 0.f;
  for (int j = t; j < 2 * DK; j += 256) {
    float s = ph1_b[j];
    for (int d = 0; d < DK; ++d) s += feat[d] * ph1_w[j * DK + d];
    float sp = (s > 20.f) ? s : log1pf(__expf(s));
    p += sp * ph2_w[j];
  }
  red[t] = p;
  __syncthreads();
  for (int off = 128; off > 0; off >>= 1) {
    if (t < off) red[t] += red[t + off];
    __syncthreads();
  }
  if (t == 0) out[b] = red[0] + ph2_b[0];
}

// ---------------------------------------------------------------------------
// Launcher
// ---------------------------------------------------------------------------
extern "C" void kernel_launch(void* const* d_in, const int* in_sizes, int n_in,
                              void* d_out, int out_size, void* d_ws, size_t ws_size,
                              hipStream_t stream) {
  const int*   x_type     = (const int*)  d_in[0];
  const float* x_feat     = (const float*)d_in[1];
  const int*   edge_index = (const int*)  d_in[2];
  const float* edge_attr  = (const float*)d_in[3];
  const int*   metal_type = (const int*)  d_in[4];
  const float* metal_feat = (const float*)d_in[5];
  const float* x_emb      = (const float*)d_in[6];
  const float* x_weight   = (const float*)d_in[7];
  const float* ew1        = (const float*)d_in[8];
  const float* mlp_w      = (const float*)d_in[9];
  const float* mlp_b      = (const float*)d_in[10];
  const float* bn_g       = (const float*)d_in[11];
  const float* bn_b       = (const float*)d_in[12];
  const float* bn_rm      = (const float*)d_in[13];
  const float* bn_rv      = (const float*)d_in[14];
  const float* me1_w      = (const float*)d_in[15];
  const float* me1_b      = (const float*)d_in[16];
  const float* me2        = (const float*)d_in[17];
  const float* wq         = (const float*)d_in[18];
  const float* wk         = (const float*)d_in[19];
  const float* wv         = (const float*)d_in[20];
  const float* ml_w       = (const float*)d_in[21];
  const float* ml_b       = (const float*)d_in[22];
  const float* lg_w       = (const float*)d_in[23];
  const float* lg_b       = (const float*)d_in[24];
  const float* ph1_w      = (const float*)d_in[25];
  const float* ph1_b      = (const float*)d_in[26];
  const float* ph2_w      = (const float*)d_in[27];
  const float* ph2_b      = (const float*)d_in[28];

  // workspace layout (all offsets 32B-aligned)
  const size_t H_BYTES = (size_t)NNODES * EMB * sizeof(float);   // 60 MB
  char* ws = (char*)d_ws;
  float*          h    = (float*)(ws);
  float*          hnew = (float*)(ws + H_BYTES);
  unsigned short* wbf  = (unsigned short*)(ws + 2 * H_BYTES);    // 5*304*320 bf16

  // 1) weight pre-conversion to padded bf16
  {
    int n = NLAYERS * NP * KPAD;
    cvt_w_kernel<<<(n + 255) / 256, 256, 0, stream>>>(mlp_w, wbf);
  }
  // 2) node init
  {
    int n = NNODES * EMB;
    init_h_kernel<<<(n + 255) / 256, 256, 0, stream>>>(x_type, x_feat, x_emb,
                                                       x_weight, h);
  }
  // 3) message-passing layers
  for (int l = 0; l < NLAYERS; ++l) {
    hipMemsetAsync(hnew, 0, H_BYTES, stream);
    msg_gemm_scatter_kernel<<<NMSG / ROWS, 128, 0, stream>>>(
        h, hnew,
        wbf + (size_t)l * NP * KPAD,
        mlp_b + l * EMB,
        edge_index, edge_attr,
        ew1 + l * 2 * EMB, ew1 + l * 2 * EMB + EMB);
    int n = NNODES * EMB;
    bn_kernel<<<(n + 255) / 256, 256, 0, stream>>>(
        hnew, h,
        bn_g + l * EMB, bn_b + l * EMB, bn_rm + l * EMB, bn_rv + l * EMB,
        (l < NLAYERS - 1) ? 1 : 0);
  }
  // 4) fused attention / pooling head
  head_kernel<<<NB, 256, 0, stream>>>(h, metal_type, metal_feat,
                                      me1_w, me1_b, me2,
                                      wq, wk, wv, ml_w, ml_b, lg_w, lg_b,
                                      ph1_w, ph1_b, ph2_w, ph2_b,
                                      (float*)d_out);
}
